// FastWeightAttention_90211493085565
// MI455X (gfx1250) — matrix operand order repaired and checked by reading, not compiled, not run on hardware
//
#include <hip/hip_runtime.h>
#include <hip/hip_bf16.h>

#define BB 4
#define NN 1024
#define DIMD 1024
#define HH 16
#define DH 64
#define ATT_SCALE 0.125f
#define MAXLR 0.01f

typedef __bf16 bf16;
typedef __attribute__((ext_vector_type(16))) __bf16 v16bf;
typedef __attribute__((ext_vector_type(8))) float v8f;

__device__ __forceinline__ int lane_id() { return threadIdx.x & 31; }
__device__ __forceinline__ int wave_id() { return threadIdx.x >> 5; }

__device__ __forceinline__ v8f wmma_bf(v16bf a, v16bf b, v8f c) {
  return __builtin_amdgcn_wmma_f32_16x16x32_bf16(false, a, false, b, (short)0, c,
                                                 false, false);
}

// ---- gfx1250 async global->LDS copy (ASYNCcnt) ----------------------------
// Flat LDS address: low 32 bits are the DS offset (ISA 10.2 aperture rules).
__device__ __forceinline__ void async_g2l_b128(const bf16* lds,
                                               const bf16* g) {
  unsigned l = (unsigned)(size_t)lds;
  unsigned long long a = (unsigned long long)(size_t)g;
  asm volatile("global_load_async_to_lds_b128 %0, %1, off" ::"v"(l), "v"(a)
               : "memory");
}
__device__ __forceinline__ void wait_async0() {
#if __has_builtin(__builtin_amdgcn_s_wait_asynccnt)
  __builtin_amdgcn_s_wait_asynccnt(0);
#else
  asm volatile("s_wait_asynccnt 0" ::: "memory");
#endif
}

// ---- WMMA fragment loaders (all contiguous -> ds_load_b128) ---------------
// A operand from [M][K]-stored LDS tile. lane L: M = L%16, K-half = (L/16)*8.
__device__ __forceinline__ v16bf frag_a(const bf16* s, int ld, int m0, int k0) {
  const int L = lane_id();
  const bf16* p = s + (size_t)(m0 + (L & 15)) * ld + k0 + ((L >> 4) << 3);
  union { uint4 q[2]; v16bf v; } u;
  u.q[0] = *(const uint4*)p;
  u.q[1] = *(const uint4*)(p + 16);
  return u.v;
}
// B operand from [N][K]-stored LDS tile. lane L: N = L%16, K-half = (L/16)*8.
__device__ __forceinline__ v16bf frag_bT(const bf16* s, int ld, int k0, int n0) {
  const int L = lane_id();
  const bf16* p = s + (size_t)(n0 + (L & 15)) * ld + k0 + ((L >> 4) << 3);
  union { uint4 q[2]; v16bf v; } u;
  u.q[0] = *(const uint4*)p;
  u.q[1] = *(const uint4*)(p + 16);
  return u.v;
}

// Stage a 64x64 bf16 tile from global into LDS (ld = 72 both layouts).
// Normal-only layout uses the async global->LDS engine; transposing layouts
// need the data in VGPRs anyway, so they load + store/scatter.
template <bool WN, bool WT>
__device__ __forceinline__ void stage_tile(const bf16* g, int rs, bf16* dN,
                                           bf16* dT) {
  const int tid = threadIdx.x;
#pragma unroll
  for (int c = 0; c < 2; ++c) {
    int idx = tid + c * 256;
    int r = idx >> 3, c0 = (idx & 7) << 3;
    const bf16* src = g + (size_t)r * rs + c0;
    if (WN && !WT) {
      async_g2l_b128(dN + (size_t)r * 72 + c0, src);
    } else {
      uint4 d = *(const uint4*)src;
      if (WN) *(uint4*)(dN + (size_t)r * 72 + c0) = d;
      if (WT) {
        const bf16* e = (const bf16*)&d;
#pragma unroll
        for (int j = 0; j < 8; ++j) dT[(size_t)(c0 + j) * 72 + r] = e[j];
      }
    }
  }
}

// ---------------- generic 64x64-tile GEMM (bf16 in, f32 accum) -------------
struct GemmSmem {
  bf16 As[64][72];  // A(m,k), K-chunk 64
  bf16 Bs[64][72];  // B(k,n) stored [n][k]
};

template <bool TA, bool TB>
__device__ void gemm64(const bf16* A, int lda, const bf16* B, int ldb,
                       float* Cf, bf16* Cbf, int ldc, int M, int Nn, int K,
                       int m0, int n0, GemmSmem* sm) {
  const int tid = threadIdx.x;
  const int w = wave_id();
  const int mt = ((2 * w) >> 2) << 4;
  const int nt0 = ((2 * w) & 3) << 4;
  const int nt1 = nt0 + 16;
  v8f acc0 = {}, acc1 = {};
  for (int k0 = 0; k0 < K; k0 += 64) {
    __syncthreads();
    if (!TA) {  // A[m][k] global: async copy straight into LDS.
      // (Ragged M: the over-read row only feeds C rows the epilogue skips.)
#pragma unroll
      for (int c = 0; c < 2; ++c) {
        int idx = tid + c * 256;
        int r = idx >> 3, c0 = (idx & 7) << 3;
        async_g2l_b128(&sm->As[r][c0], A + (size_t)(m0 + r) * lda + k0 + c0);
      }
    } else {  // A[k][m] global, vector along m, scatter to As[m][k]
#pragma unroll
      for (int c = 0; c < 2; ++c) {
        int idx = tid + c * 256;
        int k = idx >> 3, mv = (idx & 7) << 3;
        uint4 d = *(const uint4*)(A + (size_t)(k0 + k) * lda + m0 + mv);
        const bf16* e = (const bf16*)&d;
#pragma unroll
        for (int j = 0; j < 8; ++j) sm->As[mv + j][k] = e[j];
      }
    }
    if (TB) {  // B[n][k] global: async copy straight into LDS
#pragma unroll
      for (int c = 0; c < 2; ++c) {
        int idx = tid + c * 256;
        int n = idx >> 3, c0 = (idx & 7) << 3;
        async_g2l_b128(&sm->Bs[n][c0], B + (size_t)(n0 + n) * ldb + k0 + c0);
      }
    } else {  // B[k][n] global, vector along n, scatter to Bs[n][k]
#pragma unroll
      for (int c = 0; c < 2; ++c) {
        int idx = tid + c * 256;
        int k = idx >> 3, nv = (idx & 7) << 3;
        uint4 d = *(const uint4*)(B + (size_t)(k0 + k) * ldb + n0 + nv);
        const bf16* e = (const bf16*)&d;
#pragma unroll
        for (int j = 0; j < 8; ++j) sm->Bs[nv + j][k] = e[j];
      }
    }
    wait_async0();
    __syncthreads();
#pragma unroll
    for (int kk = 0; kk < 64; kk += 32) {
      v16bf af = frag_a(&sm->As[0][0], 72, mt, kk);
      acc0 = wmma_bf(af, frag_bT(&sm->Bs[0][0], 72, kk, nt0), acc0);
      acc1 = wmma_bf(af, frag_bT(&sm->Bs[0][0], 72, kk, nt1), acc1);
    }
  }
  const int L = lane_id();
  const int er = (L >> 4) << 3;
  const int ec = L & 15;
#pragma unroll
  for (int r = 0; r < 8; ++r) {
    int gm = m0 + mt + er + r;
    if (gm < M) {
      int gn0 = n0 + nt0 + ec, gn1 = n0 + nt1 + ec;
      if (gn0 < Nn) {
        if (Cf) Cf[(size_t)gm * ldc + gn0] = acc0[r];
        if (Cbf) Cbf[(size_t)gm * ldc + gn0] = (bf16)acc0[r];
      }
      if (gn1 < Nn) {
        if (Cf) Cf[(size_t)gm * ldc + gn1] = acc1[r];
        if (Cbf) Cbf[(size_t)gm * ldc + gn1] = (bf16)acc1[r];
      }
    }
  }
}

// ---------------- small elementwise / reduction kernels --------------------
__global__ __launch_bounds__(256) void k_f2bf(const float* x, bf16* y, int n) {
  const float4* x4 = (const float4*)x;
  int stride = gridDim.x * 256;
  for (int i = blockIdx.x * 256 + threadIdx.x; i < n / 8; i += stride) {
    float4 a = x4[2 * i], b = x4[2 * i + 1];
    alignas(16) bf16 t[8] = {(bf16)a.x, (bf16)a.y, (bf16)a.z, (bf16)a.w,
                             (bf16)b.x, (bf16)b.y, (bf16)b.z, (bf16)b.w};
    *(uint4*)(y + (size_t)i * 8) = *(const uint4*)t;
  }
}

__global__ __launch_bounds__(256) void k_rmsnorm(const float* tokens,
                                                 const float* rmsw, float* t,
                                                 bf16* tbf) {
  __shared__ float red[256];
  __shared__ float s_scale;
  const size_t row = blockIdx.x;
  const int tid = threadIdx.x;
  const float4* x4 = (const float4*)(tokens + row * DIMD);
  float4 v = x4[tid];
  red[tid] = v.x * v.x + v.y * v.y + v.z * v.z + v.w * v.w;
  __syncthreads();
  for (int o = 128; o > 0; o >>= 1) {
    if (tid < o) red[tid] += red[tid + o];
    __syncthreads();
  }
  if (tid == 0) s_scale = rsqrtf(red[0] / (float)DIMD + 1.1920929e-7f);
  __syncthreads();
  float sc = s_scale;
  float4 wv = ((const float4*)rmsw)[tid];
  float4 o;
  o.x = v.x * sc * wv.x;
  o.y = v.y * sc * wv.y;
  o.z = v.z * sc * wv.z;
  o.w = v.w * sc * wv.w;
  ((float4*)(t + row * DIMD))[tid] = o;
  alignas(8) bf16 tb[4] = {(bf16)o.x, (bf16)o.y, (bf16)o.z, (bf16)o.w};
  *(uint2*)(tbf + row * DIMD + tid * 4) = *(const uint2*)tb;
}

// target = LN(tv); lr = sigmoid(tok . lr_w)*MAX_LR; err = (target-pred)*lr
// (row NN-1 of error forced to zero: zero-pads all downstream GEMMs)
__global__ __launch_bounds__(256) void k_error(const float* tv, const float* t,
                                               const float* pred,
                                               const float* lrw, bf16* errbf) {
  const size_t row = blockIdx.x;
  const int m = (int)(row % NN);
  const int tid = threadIdx.x;
  if (m == NN - 1) {
    uint2 z;
    z.x = 0u;
    z.y = 0u;
    *(uint2*)(errbf + row * DIMD + tid * 4) = z;
    return;
  }
  __shared__ float r0[256], r1[256], r2[256];
  float4 v = ((const float4*)(tv + row * DIMD))[tid];
  float4 tr = ((const float4*)(t + row * DIMD))[tid];
  float4 lw = ((const float4*)lrw)[tid];
  r0[tid] = v.x + v.y + v.z + v.w;
  r1[tid] = v.x * v.x + v.y * v.y + v.z * v.z + v.w * v.w;
  r2[tid] = tr.x * lw.x + tr.y * lw.y + tr.z * lw.z + tr.w * lw.w;
  __syncthreads();
  for (int o = 128; o > 0; o >>= 1) {
    if (tid < o) {
      r0[tid] += r0[tid + o];
      r1[tid] += r1[tid + o];
      r2[tid] += r2[tid + o];
    }
    __syncthreads();
  }
  float mu = r0[0] / (float)DIMD;
  float var = r1[0] / (float)DIMD - mu * mu;
  float rstd = rsqrtf(var + 1e-5f);
  float lr = MAXLR / (1.f + expf(-r2[0]));
  float4 pp = ((const float4*)(pred + row * DIMD))[tid];
  alignas(8) bf16 eb[4] = {(bf16)(((v.x - mu) * rstd - pp.x) * lr),
                           (bf16)(((v.y - mu) * rstd - pp.y) * lr),
                           (bf16)(((v.z - mu) * rstd - pp.z) * lr),
                           (bf16)(((v.w - mu) * rstd - pp.w) * lr)};
  *(uint2*)(errbf + row * DIMD + tid * 4) = *(const uint2*)eb;
}

// delta[b,n,h] = sum_e dout*out   (layout (B*N, H))
__global__ __launch_bounds__(256) void k_delta(const float* out,
                                               const float* dout,
                                               float* delta) {
  __shared__ float red[256];
  const size_t row = blockIdx.x;
  const int tid = threadIdx.x;
  const int h = tid >> 4, l = tid & 15;
  float4 a = ((const float4*)(out + row * 1024 + h * DH))[l];
  float4 b = ((const float4*)(dout + row * 1024 + h * DH))[l];
  red[tid] = a.x * b.x + a.y * b.y + a.z * b.z + a.w * b.w;
  __syncthreads();
  for (int o2 = 8; o2 > 0; o2 >>= 1) {
    if (l < o2) red[tid] += red[tid + o2];
    __syncthreads();
  }
  if (l == 0) delta[row * HH + h] = red[tid];
}

// ---------------- flash-style attention forward ----------------------------
__global__ __launch_bounds__(256) void k_attn_fwd(const bf16* qbf,
                                                  const bf16* kbf,
                                                  const bf16* vbf, float* outf,
                                                  bf16* outbf, float* Lse) {
  __shared__ bf16 Qs[64][72], Ks[64][72], VsT[64][72], Ps[64][72];
  __shared__ float Sf[64][68];
  __shared__ float m_s[64], l_s[64], sc_s[64];
  const int ib = blockIdx.x, h = blockIdx.y, b = blockIdx.z;
  const int tid = threadIdx.x;
  const size_t base = (size_t)b * NN * 1024 + (size_t)h * DH;
  stage_tile<true, false>(qbf + base + (size_t)(ib * 64) * 1024, 1024,
                          &Qs[0][0], nullptr);
  if (tid < 64) {
    m_s[tid] = -1e30f;
    l_s[tid] = 0.f;
  }
  const int w = wave_id();
  const int mt = ((2 * w) >> 2) << 4;
  const int nt0 = ((2 * w) & 3) << 4;
  const int nt1 = nt0 + 16;
  const int L = lane_id();
  const int er = (L >> 4) << 3;
  const int ec = L & 15;
  v8f o0 = {}, o1 = {};
  for (int jb = 0; jb <= ib; ++jb) {
    __syncthreads();
    stage_tile<true, false>(kbf + base + (size_t)(jb * 64) * 1024, 1024,
                            &Ks[0][0], nullptr);
    stage_tile<false, true>(vbf + base + (size_t)(jb * 64) * 1024, 1024,
                            nullptr, &VsT[0][0]);
    if (jb < ib) {  // prefetch next K/V tiles into cache
      const bf16* nk = kbf + base + (size_t)((jb + 1) * 64) * 1024;
      const bf16* nv = vbf + base + (size_t)((jb + 1) * 64) * 1024;
      int r = tid >> 2, c0 = (tid & 3) << 8;
      __builtin_prefetch(nk + (size_t)r * 1024 + c0, 0, 1);
      __builtin_prefetch(nv + (size_t)r * 1024 + c0, 0, 1);
    }
    wait_async0();
    __syncthreads();
    {  // S = Q * K^T  (Ks natural [j][e] is exactly the [N][K] layout)
      v8f s0 = {}, s1 = {};
#pragma unroll
      for (int kk = 0; kk < DH; kk += 32) {
        v16bf af = frag_a(&Qs[0][0], 72, mt, kk);
        s0 = wmma_bf(af, frag_bT(&Ks[0][0], 72, kk, nt0), s0);
        s1 = wmma_bf(af, frag_bT(&Ks[0][0], 72, kk, nt1), s1);
      }
#pragma unroll
      for (int r = 0; r < 8; ++r) {
        Sf[mt + er + r][nt0 + ec] = s0[r];
        Sf[mt + er + r][nt1 + ec] = s1[r];
      }
    }
    __syncthreads();
    if (tid < 64) {  // online softmax, one thread per query row
      int r = tid;
      int grow = ib * 64 + r;
      float mo = m_s[r];
      float mx = mo;
      for (int c = 0; c < 64; ++c) {
        int gc = jb * 64 + c;
        float sv = (gc <= grow) ? Sf[r][c] * ATT_SCALE : -1e30f;
        Sf[r][c] = sv;
        mx = fmaxf(mx, sv);
      }
      float corr = expf(mo - mx);
      float ls = l_s[r] * corr;
      for (int c = 0; c < 64; ++c) {
        float pv = expf(Sf[r][c] - mx);
        Ps[r][c] = (bf16)pv;
        ls += pv;
      }
      m_s[r] = mx;
      l_s[r] = ls;
      sc_s[r] = corr;
    }
    __syncthreads();
#pragma unroll
    for (int r = 0; r < 8; ++r) {
      float f = sc_s[mt + er + r];
      o0[r] *= f;
      o1[r] *= f;
    }
#pragma unroll
    for (int kk = 0; kk < 64; kk += 32) {  // O += P * V   (V^T stored)
      v16bf pf = frag_a(&Ps[0][0], 72, mt, kk);
      o0 = wmma_bf(pf, frag_bT(&VsT[0][0], 72, kk, nt0), o0);
      o1 = wmma_bf(pf, frag_bT(&VsT[0][0], 72, kk, nt1), o1);
    }
  }
  __syncthreads();
#pragma unroll
  for (int r = 0; r < 8; ++r) {
    int row = mt + er + r;
    float inv = 1.f / l_s[row];
    size_t g = base + (size_t)(ib * 64 + row) * 1024;
    float a0 = o0[r] * inv, a1 = o1[r] * inv;
    outf[g + nt0 + ec] = a0;
    outbf[g + nt0 + ec] = (bf16)a0;
    outf[g + nt1 + ec] = a1;
    outbf[g + nt1 + ec] = (bf16)a1;
  }
  if (tid < 64)
    Lse[((size_t)b * NN + ib * 64 + tid) * HH + h] = m_s[tid] + logf(l_s[tid]);
}

// ---------------- attention backward: dK/dV (per column block) -------------
__global__ __launch_bounds__(256) void k_attn_dkdv(
    const bf16* qbf, const bf16* kbf, const bf16* vbf, const bf16* dobf,
    const float* Lse, const float* delta, bf16* dkbf, bf16* dvbf) {
  __shared__ bf16 Ks[64][72], Vs[64][72], Qs[64][72], QsT[64][72], Ds[64][72],
      DsT[64][72], PsT[64][72], dSsT[64][72];
  __shared__ float Sf[64][68];
  __shared__ float Lrow[64], Drow[64];
  const int jb = blockIdx.x, h = blockIdx.y, b = blockIdx.z;
  const int tid = threadIdx.x;
  const size_t base = (size_t)b * NN * 1024 + (size_t)h * DH;
  stage_tile<true, false>(kbf + base + (size_t)(jb * 64) * 1024, 1024,
                          &Ks[0][0], nullptr);
  stage_tile<true, false>(vbf + base + (size_t)(jb * 64) * 1024, 1024,
                          &Vs[0][0], nullptr);
  const int w = wave_id();
  const int mt = ((2 * w) >> 2) << 4;
  const int nt0 = ((2 * w) & 3) << 4;
  const int nt1 = nt0 + 16;
  const int L = lane_id();
  const int er = (L >> 4) << 3;
  const int ec = L & 15;
  v8f dv0 = {}, dv1 = {}, dk0 = {}, dk1 = {};
  for (int ib = jb; ib < NN / 64; ++ib) {
    __syncthreads();
    stage_tile<true, true>(qbf + base + (size_t)(ib * 64) * 1024, 1024,
                           &Qs[0][0], &QsT[0][0]);
    stage_tile<true, true>(dobf + base + (size_t)(ib * 64) * 1024, 1024,
                           &Ds[0][0], &DsT[0][0]);
    if (tid < 64) {
      size_t rg = ((size_t)b * NN + ib * 64 + tid) * HH + h;
      Lrow[tid] = Lse[rg];
      Drow[tid] = delta[rg];
    }
    if (ib + 1 < NN / 64) {
      const bf16* nq = qbf + base + (size_t)((ib + 1) * 64) * 1024;
      const bf16* nd = dobf + base + (size_t)((ib + 1) * 64) * 1024;
      int r = tid >> 2, c0 = (tid & 3) << 8;
      __builtin_prefetch(nq + (size_t)r * 1024 + c0, 0, 1);
      __builtin_prefetch(nd + (size_t)r * 1024 + c0, 0, 1);
    }
    wait_async0();
    __syncthreads();
    {  // S = Qi * Kj^T
      v8f s0 = {}, s1 = {};
#pragma unroll
      for (int kk = 0; kk < DH; kk += 32) {
        v16bf af = frag_a(&Qs[0][0], 72, mt, kk);
        s0 = wmma_bf(af, frag_bT(&Ks[0][0], 72, kk, nt0), s0);
        s1 = wmma_bf(af, frag_bT(&Ks[0][0], 72, kk, nt1), s1);
      }
#pragma unroll
      for (int r = 0; r < 8; ++r) {
        Sf[mt + er + r][nt0 + ec] = s0[r];
        Sf[mt + er + r][nt1 + ec] = s1[r];
      }
    }
    __syncthreads();
#pragma unroll 4
    for (int t2 = 0; t2 < 16; ++t2) {  // P^T = exp(S*scale - L)
      int i = tid + t2 * 256;
      int r = i >> 6, c = i & 63;
      int gi = ib * 64 + r, gj = jb * 64 + c;
      float pv = (gj <= gi) ? expf(Sf[r][c] * ATT_SCALE - Lrow[r]) : 0.f;
      PsT[c][r] = (bf16)pv;
    }
    __syncthreads();
    {  // dP = dO * V^T -> Sf   (Vs natural [j][e] is the [N][K] layout)
      v8f s0 = {}, s1 = {};
#pragma unroll
      for (int kk = 0; kk < DH; kk += 32) {
        v16bf af = frag_a(&Ds[0][0], 72, mt, kk);
        s0 = wmma_bf(af, frag_bT(&Vs[0][0], 72, kk, nt0), s0);
        s1 = wmma_bf(af, frag_bT(&Vs[0][0], 72, kk, nt1), s1);
      }
#pragma unroll
      for (int r = 0; r < 8; ++r) {
        Sf[mt + er + r][nt0 + ec] = s0[r];
        Sf[mt + er + r][nt1 + ec] = s1[r];
      }
    }
    __syncthreads();
#pragma unroll 4
    for (int t2 = 0; t2 < 16; ++t2) {  // dS^T = scale*P*(dP-delta)
      int i = tid + t2 * 256;
      int r = i >> 6, c = i & 63;
      float pv = (float)PsT[c][r];
      dSsT[c][r] = (bf16)(ATT_SCALE * pv * (Sf[r][c] - Drow[r]));
    }
    __syncthreads();
#pragma unroll
    for (int kk = 0; kk < 64; kk += 32) {  // dv += P^T*dO ; dk += dS^T*Q
      v16bf ap = frag_a(&PsT[0][0], 72, mt, kk);
      dv0 = wmma_bf(ap, frag_bT(&DsT[0][0], 72, kk, nt0), dv0);
      dv1 = wmma_bf(ap, frag_bT(&DsT[0][0], 72, kk, nt1), dv1);
      v16bf as = frag_a(&dSsT[0][0], 72, mt, kk);
      dk0 = wmma_bf(as, frag_bT(&QsT[0][0], 72, kk, nt0), dk0);
      dk1 = wmma_bf(as, frag_bT(&QsT[0][0], 72, kk, nt1), dk1);
    }
  }
#pragma unroll
  for (int r = 0; r < 8; ++r) {
    size_t g = base + (size_t)(jb * 64 + mt + er + r) * 1024;
    dvbf[g + nt0 + ec] = (bf16)dv0[r];
    dvbf[g + nt1 + ec] = (bf16)dv1[r];
    dkbf[g + nt0 + ec] = (bf16)dk0[r];
    dkbf[g + nt1 + ec] = (bf16)dk1[r];
  }
}

// ---------------- attention backward: dQ (per row block) -------------------
__global__ __launch_bounds__(256) void k_attn_dq(const bf16* qbf,
                                                 const bf16* kbf,
                                                 const bf16* vbf,
                                                 const bf16* dobf,
                                                 const float* Lse,
                                                 const float* delta,
                                                 bf16* dqbf) {
  __shared__ bf16 Qs[64][72], Ds[64][72], Ks[64][72], KsT[64][72], Vs[64][72],
      Ps[64][72], dSs[64][72];
  __shared__ float Sf[64][68];
  __shared__ float Lrow[64], Drow[64];
  const int ib = blockIdx.x, h = blockIdx.y, b = blockIdx.z;
  const int tid = threadIdx.x;
  const size_t base = (size_t)b * NN * 1024 + (size_t)h * DH;
  stage_tile<true, false>(qbf + base + (size_t)(ib * 64) * 1024, 1024,
                          &Qs[0][0], nullptr);
  stage_tile<true, false>(dobf + base + (size_t)(ib * 64) * 1024, 1024,
                          &Ds[0][0], nullptr);
  if (tid < 64) {
    size_t rg = ((size_t)b * NN + ib * 64 + tid) * HH + h;
    Lrow[tid] = Lse[rg];
    Drow[tid] = delta[rg];
  }
  const int w = wave_id();
  const int mt = ((2 * w) >> 2) << 4;
  const int nt0 = ((2 * w) & 3) << 4;
  const int nt1 = nt0 + 16;
  const int L = lane_id();
  const int er = (L >> 4) << 3;
  const int ec = L & 15;
  v8f dq0 = {}, dq1 = {};
  for (int jb = 0; jb <= ib; ++jb) {
    __syncthreads();
    stage_tile<true, true>(kbf + base + (size_t)(jb * 64) * 1024, 1024,
                           &Ks[0][0], &KsT[0][0]);
    stage_tile<true, false>(vbf + base + (size_t)(jb * 64) * 1024, 1024,
                            &Vs[0][0], nullptr);
    if (jb < ib) {
      const bf16* nk = kbf + base + (size_t)((jb + 1) * 64) * 1024;
      const bf16* nv = vbf + base + (size_t)((jb + 1) * 64) * 1024;
      int r = tid >> 2, c0 = (tid & 3) << 8;
      __builtin_prefetch(nk + (size_t)r * 1024 + c0, 0, 1);
      __builtin_prefetch(nv + (size_t)r * 1024 + c0, 0, 1);
    }
    wait_async0();
    __syncthreads();
    {  // S
      v8f s0 = {}, s1 = {};
#pragma unroll
      for (int kk = 0; kk < DH; kk += 32) {
        v16bf af = frag_a(&Qs[0][0], 72, mt, kk);
        s0 = wmma_bf(af, frag_bT(&Ks[0][0], 72, kk, nt0), s0);
        s1 = wmma_bf(af, frag_bT(&Ks[0][0], 72, kk, nt1), s1);
      }
#pragma unroll
      for (int r = 0; r < 8; ++r) {
        Sf[mt + er + r][nt0 + ec] = s0[r];
        Sf[mt + er + r][nt1 + ec] = s1[r];
      }
    }
    __syncthreads();
#pragma unroll 4
    for (int t2 = 0; t2 < 16; ++t2) {
      int i = tid + t2 * 256;
      int r = i >> 6, c = i & 63;
      int gi = ib * 64 + r, gj = jb * 64 + c;
      float pv = (gj <= gi) ? expf(Sf[r][c] * ATT_SCALE - Lrow[r]) : 0.f;
      Ps[r][c] = (bf16)pv;
    }
    __syncthreads();
    {  // dP
      v8f s0 = {}, s1 = {};
#pragma unroll
      for (int kk = 0; kk < DH; kk += 32) {
        v16bf af = frag_a(&Ds[0][0], 72, mt, kk);
        s0 = wmma_bf(af, frag_bT(&Vs[0][0], 72, kk, nt0), s0);
        s1 = wmma_bf(af, frag_bT(&Vs[0][0], 72, kk, nt1), s1);
      }
#pragma unroll
      for (int r = 0; r < 8; ++r) {
        Sf[mt + er + r][nt0 + ec] = s0[r];
        Sf[mt + er + r][nt1 + ec] = s1[r];
      }
    }
    __syncthreads();
#pragma unroll 4
    for (int t2 = 0; t2 < 16; ++t2) {
      int i = tid + t2 * 256;
      int r = i >> 6, c = i & 63;
      float pv = (float)Ps[r][c];
      dSs[r][c] = (bf16)(ATT_SCALE * pv * (Sf[r][c] - Drow[r]));
    }
    __syncthreads();
#pragma unroll
    for (int kk = 0; kk < 64; kk += 32) {  // dq += dS * K   (K^T stored)
      v16bf as = frag_a(&dSs[0][0], 72, mt, kk);
      dq0 = wmma_bf(as, frag_bT(&KsT[0][0], 72, kk, nt0), dq0);
      dq1 = wmma_bf(as, frag_bT(&KsT[0][0], 72, kk, nt1), dq1);
    }
  }
#pragma unroll
  for (int r = 0; r < 8; ++r) {
    size_t g = base + (size_t)(ib * 64 + mt + er + r) * 1024;
    dqbf[g + nt0 + ec] = (bf16)dq0[r];
    dqbf[g + nt1 + ec] = (bf16)dq1[r];
  }
}

// ---------------- Newton-Schulz (5 iters) on 64x1024 matrices in LDS -------
__global__ __launch_bounds__(256) void k_ns(const float* dwv_raw,
                                            const float* dwo_raw,
                                            float* out_dwv, float* out_dwo) {
  const float acoef = 3.4445f, bcoef = -4.775f, ccoef = 2.0315f;
  const int which = blockIdx.x;
  const int h = blockIdx.y, b = blockIdx.z;
  const int mat = b * HH + h;
  const int tid = threadIdx.x;
  extern __shared__ char sm8[];
  bf16* X = (bf16*)sm8;                           // [64][1032]
  bf16* XT = X + (size_t)64 * 1032;               // [1024][72]
  float* Af = (float*)(XT + (size_t)1024 * 72);   // [64][68]
  bf16* Abf = (bf16*)(Af + (size_t)64 * 68);      // [64][72]
  bf16* AbfT = Abf + (size_t)64 * 72;             // [64][72]
  bf16* Bmbf = AbfT + (size_t)64 * 72;            // [64][72]
  float* red = (float*)(Bmbf + (size_t)64 * 72);  // [256]
  const float* G = (which == 0) ? dwv_raw + (size_t)mat * DIMD * DH
                                : dwo_raw + (size_t)mat * DH * DIMD;
  float ss = 0.f;  // Frobenius norm
  const float4* G4 = (const float4*)G;
#pragma unroll 4
  for (int t2 = 0; t2 < 64; ++t2) {
    float4 v = G4[tid + t2 * 256];
    ss += v.x * v.x + v.y * v.y + v.z * v.z + v.w * v.w;
  }
  red[tid] = ss;
  __syncthreads();
  for (int o = 128; o > 0; o >>= 1) {
    if (tid < o) red[tid] += red[tid + o];
    __syncthreads();
  }
  const float scl = 1.f / (sqrtf(red[0]) + 1e-7f);
  if (which == 0) {  // G is (1024 x 64): X = G^T
    for (int t2 = 0; t2 < 32; ++t2) {
      int idx = tid + t2 * 256;
      int i = idx >> 3, e0 = (idx & 7) << 3;
      alignas(16) bf16 tmp[8];
#pragma unroll
      for (int j = 0; j < 8; ++j)
        tmp[j] = (bf16)(G[(size_t)i * DH + e0 + j] * scl);
      *(uint4*)(XT + (size_t)i * 72 + e0) = *(const uint4*)tmp;
#pragma unroll
      for (int j = 0; j < 8; ++j) X[(size_t)(e0 + j) * 1032 + i] = tmp[j];
    }
  } else {  // G is (64 x 1024): X = G
    for (int t2 = 0; t2 < 32; ++t2) {
      int idx = tid + t2 * 256;
      int e = idx >> 7, i0 = (idx & 127) << 3;
      alignas(16) bf16 tmp[8];
#pragma unroll
      for (int j = 0; j < 8; ++j)
        tmp[j] = (bf16)(G[(size_t)e * DIMD + i0 + j] * scl);
      *(uint4*)(X + (size_t)e * 1032 + i0) = *(const uint4*)tmp;
#pragma unroll
      for (int j = 0; j < 8; ++j) XT[(size_t)(i0 + j) * 72 + e] = tmp[j];
    }
  }
  const int w = wave_id();
  const int mt = ((2 * w) >> 2) << 4;
  const int nt0 = ((2 * w) & 3) << 4;
  const int nt1 = nt0 + 16;
  const int L = lane_id();
  const int er = (L >> 4) << 3;
  const int ec = L & 15;
  for (int it = 0; it < 5; ++it) {
    __syncthreads();
    {  // A = X * X^T  (X natural layout serves both operands)
      v8f a0 = {}, a1 = {};
      for (int kk = 0; kk < 1024; kk += 32) {
        v16bf af = frag_a(X, 1032, mt, kk);
        a0 = wmma_bf(af, frag_bT(X, 1032, kk, nt0), a0);
        a1 = wmma_bf(af, frag_bT(X, 1032, kk, nt1), a1);
      }
#pragma unroll
      for (int r = 0; r < 8; ++r) {
        int m = mt + er + r;
        Af[m * 68 + nt0 + ec] = a0[r];
        Abf[m * 72 + nt0 + ec] = (bf16)a0[r];
        AbfT[(nt0 + ec) * 72 + m] = (bf16)a0[r];
        Af[m * 68 + nt1 + ec] = a1[r];
        Abf[m * 72 + nt1 + ec] = (bf16)a1[r];
        AbfT[(nt1 + ec) * 72 + m] = (bf16)a1[r];
      }
    }
    __syncthreads();
    {  // Bm = b*A + c*(A*A)
      v8f c0 = {}, c1 = {};
#pragma unroll
      for (int kk = 0; kk < 64; kk += 32) {
        v16bf af = frag_a(Abf, 72, mt, kk);
        c0 = wmma_bf(af, frag_bT(AbfT, 72, kk, nt0), c0);
        c1 = wmma_bf(af, frag_bT(AbfT, 72, kk, nt1), c1);
      }
#pragma unroll
      for (int r = 0; r < 8; ++r) {
        int m = mt + er + r;
        Bmbf[m * 72 + nt0 + ec] =
            (bf16)(bcoef * Af[m * 68 + nt0 + ec] + ccoef * c0[r]);
        Bmbf[m * 72 + nt1 + ec] =
            (bf16)(bcoef * Af[m * 68 + nt1 + ec] + ccoef * c1[r]);
      }
    }
    __syncthreads();
    for (int pi = 0; pi < 16; ++pi) {  // X <- a*X + Bm*X, in place per panel
      const int pc = pi * 64;
      v8f x0 = {}, x1 = {};
#pragma unroll
      for (int kk = 0; kk < 64; kk += 32) {
        v16bf bmf = frag_a(Bmbf, 72, mt, kk);
        x0 = wmma_bf(bmf, frag_bT(XT, 72, kk, pc + nt0), x0);
        x1 = wmma_bf(bmf, frag_bT(XT, 72, kk, pc + nt1), x1);
      }
#pragma unroll
      for (int r = 0; r < 8; ++r) {
        int m = mt + er + r;
        x0[r] += acoef * (float)X[(size_t)m * 1032 + pc + nt0 + ec];
        x1[r] += acoef * (float)X[(size_t)m * 1032 + pc + nt1 + ec];
      }
      __syncthreads();
#pragma unroll
      for (int r = 0; r < 8; ++r) {
        int m = mt + er + r;
        int n0g = pc + nt0 + ec, n1g = pc + nt1 + ec;
        bf16 v0 = (bf16)x0[r], v1 = (bf16)x1[r];
        X[(size_t)m * 1032 + n0g] = v0;
        XT[(size_t)n0g * 72 + m] = v0;
        X[(size_t)m * 1032 + n1g] = v1;
        XT[(size_t)n1g * 72 + m] = v1;
      }
      __syncthreads();
    }
  }
  float* dst = (which == 0) ? out_dwv + (size_t)mat * DIMD * DH
                            : out_dwo + (size_t)mat * DH * DIMD;
  if (which == 0) {  // (DIM, DH): row i of dst = row i of XT
    for (int t2 = 0; t2 < 32; ++t2) {
      int idx = tid + t2 * 256;
      int i = idx >> 3, e0 = (idx & 7) << 3;
      alignas(16) bf16 tmp[8];
      *(uint4*)tmp = *(const uint4*)(XT + (size_t)i * 72 + e0);
#pragma unroll
      for (int j = 0; j < 8; ++j) dst[(size_t)i * DH + e0 + j] = (float)tmp[j];
    }
  } else {  // (DH, DIM): row e of dst = row e of X
    for (int t2 = 0; t2 < 32; ++t2) {
      int idx = tid + t2 * 256;
      int e = idx >> 7, i0 = (idx & 127) << 3;
      alignas(16) bf16 tmp[8];
      *(uint4*)tmp = *(const uint4*)(X + (size_t)e * 1032 + i0);
#pragma unroll
      for (int j = 0; j < 8; ++j)
        dst[(size_t)e * DIMD + i0 + j] = (float)tmp[j];
    }
  }
}

// ---------------- GEMM wrapper kernels -------------------------------------
__global__ __launch_bounds__(256) void k_gemm_proj(const bf16* t,
                                                   const bf16* W, bf16* obf) {
  __shared__ GemmSmem sm;
  int b = blockIdx.z / HH, h = blockIdx.z % HH;
  gemm64<false, false>(t + (size_t)b * NN * DIMD, DIMD,
                       W + (size_t)h * DIMD * DH, DH, nullptr,
                       obf + (size_t)b * NN * 1024 + h * DH, 1024, NN, DH,
                       DIMD, blockIdx.x * 64, 0, &sm);
}

__global__ __launch_bounds__(256) void k_gemm_pred(const bf16* outbf,
                                                   const bf16* wobf,
                                                   float* pred) {
  __shared__ GemmSmem sm;
  int b = blockIdx.z;
  gemm64<false, false>(outbf + (size_t)b * NN * 1024, 1024, wobf, DIMD,
                       pred + (size_t)b * NN * DIMD, nullptr, DIMD, NN, DIMD,
                       1024, blockIdx.x * 64, blockIdx.y * 64, &sm);
}

__global__ __launch_bounds__(256) void k_gemm_tv(const bf16* tbf,
                                                 const bf16* tvw, float* tv) {
  __shared__ GemmSmem sm;
  int b = blockIdx.z;
  gemm64<false, true>(tbf + (size_t)b * NN * DIMD + DIMD, DIMD, tvw, DIMD,
                      tv + (size_t)b * NN * DIMD, nullptr, DIMD, NN - 1, DIMD,
                      DIMD, blockIdx.x * 64, blockIdx.y * 64, &sm);
}

__global__ __launch_bounds__(256) void k_gemm_dout(const bf16* errbf,
                                                   const bf16* wobf,
                                                   float* doutf,
                                                   bf16* doutbf) {
  __shared__ GemmSmem sm;
  int b = blockIdx.z / HH, h = blockIdx.z % HH;
  gemm64<false, true>(errbf + (size_t)b * NN * DIMD, DIMD,
                      wobf + (size_t)h * DH * DIMD, DIMD,
                      doutf + (size_t)b * NN * 1024 + h * DH,
                      doutbf + (size_t)b * NN * 1024 + h * DH, 1024, NN, DH,
                      DIMD, blockIdx.x * 64, 0, &sm);
}

// dW[b,h,d,e] = sum_n t[b,n,d] * dX[b,n,h,e]  ->  t^T @ dX
__global__ __launch_bounds__(256) void k_gemm_dw(const bf16* tbf,
                                                 const bf16* dxbf,
                                                 float* dst) {
  __shared__ GemmSmem sm;
  int b = blockIdx.z / HH, h = blockIdx.z % HH;
  gemm64<true, false>(tbf + (size_t)b * NN * DIMD, DIMD,
                      dxbf + (size_t)b * NN * 1024 + h * DH, 1024,
                      dst + (size_t)(b * HH + h) * DIMD * DH, nullptr, DH,
                      DIMD, DH, NN, blockIdx.x * 64, 0, &sm);
}

// dwo[b,h,e,d] = sum_n out[b,n,h,e] * err[b,n,d]  ->  out^T @ err
__global__ __launch_bounds__(256) void k_gemm_dwo(const bf16* outbf,
                                                  const bf16* errbf,
                                                  float* dst) {
  __shared__ GemmSmem sm;
  int b = blockIdx.z / HH, h = blockIdx.z % HH;
  gemm64<true, false>(outbf + (size_t)b * NN * 1024 + h * DH, 1024,
                      errbf + (size_t)b * NN * DIMD, DIMD,
                      dst + (size_t)(b * HH + h) * DH * DIMD, nullptr, DIMD,
                      DH, DIMD, NN, 0, blockIdx.y * 64, &sm);
}

// ---------------- host side ------------------------------------------------
extern "C" void kernel_launch(void* const* d_in, const int* in_sizes, int n_in,
                              void* d_out, int out_size, void* d_ws,
                              size_t ws_size, hipStream_t stream) {
  (void)in_sizes;
  (void)n_in;
  (void)out_size;
  (void)ws_size;
  const float* tokens = (const float*)d_in[0];
  const float* rms_w = (const float*)d_in[1];
  const float* wq = (const float*)d_in[2];
  const float* wk = (const float*)d_in[3];
  const float* wv = (const float*)d_in[4];
  const float* wo = (const float*)d_in[5];
  const float* lr_w = (const float*)d_in[6];
  const float* tv_w = (const float*)d_in[7];

  const size_t M4 = (size_t)BB * NN * DIMD;  // 4M elements
  float* pred = (float*)d_out;
  float* dwq_o = pred + 1 * M4;
  float* dwk_o = pred + 2 * M4;
  float* dwv_o = pred + 3 * M4;
  float* dwo_o = pred + 4 * M4;

  char* p = (char*)d_ws;
  float* t_f = (float*)p;      p += M4 * 4;
  bf16* t_bf = (bf16*)p;       p += M4 * 2;
  bf16* wq_bf = (bf16*)p;      p += (size_t)HH * DIMD * DH * 2;
  bf16* wk_bf = (bf16*)p;      p += (size_t)HH * DIMD * DH * 2;
  bf16* wv_bf = (bf16*)p;      p += (size_t)HH * DIMD * DH * 2;
  bf16* wo_bf = (bf16*)p;      p += (size_t)HH * DH * DIMD * 2;
  bf16* tvw_bf = (bf16*)p;     p += (size_t)DIMD * DIMD * 2;
  bf16* q_bf = (bf16*)p;       p += M4 * 2;
  bf16* k_bf = (bf16*)p;       p += M4 * 2;
  bf16* v_bf = (bf16*)p;       p += M4 * 2;
  float* out_f = (float*)p;    p += M4 * 4;
  bf16* out_bf = (bf16*)p;     p += M4 * 2;
  float* Lse = (float*)p;      p += (size_t)BB * NN * HH * 4;
  float* tv_f = (float*)p;     p += M4 * 4;
  bf16* err_bf = (bf16*)p;     p += M4 * 2;
  float* dout_f = (float*)p;   p += M4 * 4;
  bf16* dout_bf = (bf16*)p;    p += M4 * 2;
  float* delta_f = (float*)p;  p += (size_t)BB * NN * HH * 4;
  bf16* dq_bf = (bf16*)p;      p += M4 * 2;
  bf16* dk_bf = (bf16*)p;      p += M4 * 2;
  bf16* dv_bf = (bf16*)p;      p += M4 * 2;
  float* dwv_raw = (float*)p;  p += M4 * 4;
  float* dwo_raw = (float*)p;  p += M4 * 4;

  const int WN = HH * DIMD * DH;
  k_f2bf<<<256, 256, 0, stream>>>(wq, wq_bf, WN);
  k_f2bf<<<256, 256, 0, stream>>>(wk, wk_bf, WN);
  k_f2bf<<<256, 256, 0, stream>>>(wv, wv_bf, WN);
  k_f2bf<<<256, 256, 0, stream>>>(wo, wo_bf, WN);
  k_f2bf<<<256, 256, 0, stream>>>(tv_w, tvw_bf, DIMD * DIMD);

  k_rmsnorm<<<BB * NN, 256, 0, stream>>>(tokens, rms_w, t_f, t_bf);

  dim3 gproj(NN / 64, 1, BB * HH);
  k_gemm_proj<<<gproj, 256, 0, stream>>>(t_bf, wq_bf, q_bf);
  k_gemm_proj<<<gproj, 256, 0, stream>>>(t_bf, wk_bf, k_bf);
  k_gemm_proj<<<gproj, 256, 0, stream>>>(t_bf, wv_bf, v_bf);

  k_attn_fwd<<<dim3(NN / 64, HH, BB), 256, 0, stream>>>(q_bf, k_bf, v_bf,
                                                        out_f, out_bf, Lse);

  k_gemm_pred<<<dim3(NN / 64, DIMD / 64, BB), 256, 0, stream>>>(out_bf, wo_bf,
                                                                pred);
  k_gemm_tv<<<dim3(NN / 64, DIMD / 64, BB), 256, 0, stream>>>(t_bf, tvw_bf,
                                                              tv_f);
  k_error<<<BB * NN, 256, 0, stream>>>(tv_f, t_f, pred, lr_w, err_bf);

  k_gemm_dout<<<dim3(NN / 64, 1, BB * HH), 256, 0, stream>>>(err_bf, wo_bf,
                                                             dout_f, dout_bf);
  k_delta<<<BB * NN, 256, 0, stream>>>(out_f, dout_f, delta_f);

  k_attn_dkdv<<<dim3(NN / 64, HH, BB), 256, 0, stream>>>(
      q_bf, k_bf, v_bf, dout_bf, Lse, delta_f, dk_bf, dv_bf);
  k_attn_dq<<<dim3(NN / 64, HH, BB), 256, 0, stream>>>(
      q_bf, k_bf, v_bf, dout_bf, Lse, delta_f, dq_bf);

  dim3 gdw(DIMD / 64, 1, BB * HH);
  k_gemm_dw<<<gdw, 256, 0, stream>>>(t_bf, dq_bf, dwq_o);
  k_gemm_dw<<<gdw, 256, 0, stream>>>(t_bf, dk_bf, dwk_o);
  k_gemm_dw<<<gdw, 256, 0, stream>>>(t_bf, dv_bf, dwv_raw);
  k_gemm_dwo<<<dim3(1, DIMD / 64, BB * HH), 256, 0, stream>>>(out_bf, err_bf,
                                                              dwo_raw);

  const size_t NS_SMEM = (size_t)64 * 1032 * 2      // X
                         + (size_t)1024 * 72 * 2    // XT
                         + (size_t)64 * 68 * 4      // Af
                         + (size_t)3 * 64 * 72 * 2  // Abf, AbfT, Bmbf
                         + 256 * 4;                 // red
  k_ns<<<dim3(2, HH, BB), 256, NS_SMEM, stream>>>(dwv_raw, dwo_raw, dwv_o,
                                                  dwo_o);
}
